// GAT_21260088115316
// MI455X (gfx1250) — matrix operand order repaired and checked by reading, not compiled, not run on hardware
//
#include <hip/hip_runtime.h>
#include <hip/hip_bf16.h>
#include <math.h>

typedef __attribute__((ext_vector_type(16))) _Float16 v16h;
typedef __attribute__((ext_vector_type(8)))  float    v8f;

#define NEG_SLOPE 0.2f

__device__ __forceinline__ unsigned f32_key(float f) {
    unsigned u = __float_as_uint(f);
    return (u & 0x80000000u) ? ~u : (u | 0x80000000u);   // monotonic order-preserving map
}
__device__ __forceinline__ float key_f32(unsigned k) {
    return __uint_as_float((k & 0x80000000u) ? (k & 0x7FFFFFFFu) : ~k);
}

// one-shot fp32 -> f16 weight conversion (W shared by all row tiles)
__global__ void w_to_f16(const float* __restrict__ W, _Float16* __restrict__ Wh, int n) {
    int i = blockIdx.x * blockDim.x + threadIdx.x;
    if (i < n) Wh[i] = (_Float16)W[i];
}

// ---------------------------------------------------------------------------
// WMMA GEMM:  H[N,FOUT] = X[N,FIN] * W[FOUT,FIN]^T (+ bias)
// One wave computes a 16 x FOUT tile via v_wmma_f32_16x16x32_f16.
// Weights pre-converted to f16 (packed v16h fragment loads, no cvt on B path).
// Optional fused epilogue: esrc[i] = h[i]·a_src, edst[i] = h[i]·a_dst via
// in-register partials + cross-lane shfl_xor reduction.
// ---------------------------------------------------------------------------
template<int FIN, int FOUT>
__global__ void gat_gemm_wmma(const float* __restrict__ X,
                              const _Float16* __restrict__ Wh,
                              const float* __restrict__ bias,
                              const float* __restrict__ as_,   // null -> skip dots
                              const float* __restrict__ ad_,
                              float* __restrict__ H,
                              float* __restrict__ esrc, float* __restrict__ edst,
                              int N) {
    constexpr int NT = FOUT / 16;
    const int wave = blockIdx.x * (blockDim.x >> 5) + (threadIdx.x >> 5);
    const int lane = threadIdx.x & 31;
    const int row0 = wave * 16;
    if (row0 >= N) return;                 // uniform per wave
    const int half = lane >> 4;            // 0: lanes 0-15, 1: lanes 16-31
    const int lm   = lane & 15;

    int arow = row0 + lm;
    if (arow >= N) arow = N - 1;           // clamp (duplicate loads, stores masked)
    const float* xrow = X + (long)arow * FIN;

    v8f acc[NT];
#pragma unroll
    for (int t = 0; t < NT; ++t) acc[t] = (v8f){0.f,0.f,0.f,0.f,0.f,0.f,0.f,0.f};

#pragma unroll
    for (int kb = 0; kb < FIN; kb += 32) {
        // A fragment (16x32 f16): lanes 0-15: elems 0-15 -> K 0-15
        //                         lanes 16-31: elems 0-15 -> K 16-31
        v16h a;
        const int base1 = kb + (half ? 16 : 0);
        const int base2 = kb + (half ? 24 : 8);
#pragma unroll
        for (int j = 0; j < 8; ++j) a[j]     = (_Float16)xrow[base1 + j];
#pragma unroll
        for (int j = 0; j < 8; ++j) a[8 + j] = (_Float16)xrow[base2 + j];

#pragma unroll
        for (int t = 0; t < NT; ++t) {
            // B fragment (32x16 f16): lane = column n, elems j -> K = half*16 + j
            // B[k][n] = W[n][k]; address multiple of 16 halves -> 32B aligned
            const v16h b = *(const v16h*)(Wh + (long)(t * 16 + lm) * FIN + kb + half * 16);
            acc[t] = __builtin_amdgcn_wmma_f32_16x16x32_f16(
                false, a, false, b, (short)0, acc[t], false, false);
        }
    }

    // C/D layout: VGPR r -> row row0 + 8*half + r, col = t*16 + lm
    float ps[8], pd[8];
#pragma unroll
    for (int r = 0; r < 8; ++r) { ps[r] = 0.f; pd[r] = 0.f; }

#pragma unroll
    for (int t = 0; t < NT; ++t) {
        const int col = t * 16 + lm;
        const float bv = bias ? bias[col] : 0.0f;
        const float av_s = as_ ? as_[col] : 0.0f;
        const float av_d = as_ ? ad_[col] : 0.0f;
#pragma unroll
        for (int r = 0; r < 8; ++r) {
            const int row = row0 + half * 8 + r;
            const float v = acc[t][r] + bv;
            if (row < N) H[(long)row * FOUT + col] = v;
            ps[r] += v * av_s;
            pd[r] += v * av_d;
        }
    }

    if (as_) {
        // reduce dots across the 16 lanes of each half-group
#pragma unroll
        for (int r = 0; r < 8; ++r) {
#pragma unroll
            for (int m = 8; m > 0; m >>= 1) {
                ps[r] += __shfl_xor(ps[r], m, 32);
                pd[r] += __shfl_xor(pd[r], m, 32);
            }
        }
        if (lm == 0) {
#pragma unroll
            for (int r = 0; r < 8; ++r) {
                const int row = row0 + half * 8 + r;
                if (row < N) { esrc[row] = ps[r]; edst[row] = pd[r]; }
            }
        }
    }
}

__global__ void gat_init(unsigned* __restrict__ mkey, float* __restrict__ denom,
                         float* __restrict__ acc, int N, int F) {
    long idx = (long)blockIdx.x * blockDim.x + threadIdx.x;
    if (idx < (long)N * F) acc[idx] = 0.f;
    if (idx < N) { mkey[idx] = 0u; denom[idx] = 0.f; }
}

// pass 1: segment max of leaky_relu(e_src[s]+e_dst[d]) over dst (incl. self loops)
__global__ void gat_edge_max(const long long* __restrict__ ei, int E, int N,
                             const float* __restrict__ esrc, const float* __restrict__ edst,
                             unsigned* __restrict__ mkey) {
    int e = blockIdx.x * blockDim.x + threadIdx.x;
    if (e >= E + N) return;
    int s, d;
    if (e < E) { s = (int)ei[e]; d = (int)ei[(long)E + e]; } else { s = d = e - E; }
    float v = esrc[s] + edst[d];
    v = v > 0.f ? v : NEG_SLOPE * v;
    atomicMax(&mkey[d], f32_key(v));
}

// pass 2 (one wave per edge): denom[d] += ex ; acc[d,:] += ex * h[s,:]
__global__ void gat_edge_acc(const long long* __restrict__ ei, int E, int N,
                             const float* __restrict__ esrc, const float* __restrict__ edst,
                             const unsigned* __restrict__ mkey,
                             const float* __restrict__ H, int F,
                             float* __restrict__ denom, float* __restrict__ acc) {
    int w    = (blockIdx.x * blockDim.x + threadIdx.x) >> 5;
    int lane = threadIdx.x & 31;
    if (w >= E + N) return;
    int s, d;
    if (w < E) { s = (int)ei[w]; d = (int)ei[(long)E + w]; } else { s = d = w - E; }
    float v = esrc[s] + edst[d];
    v = v > 0.f ? v : NEG_SLOPE * v;
    float ex = __expf(v - key_f32(mkey[d]));
    if (lane == 0) atomicAdd(&denom[d], ex);
    const float* hs = H + (long)s * F;
    float* ad = acc + (long)d * F;
    for (int f = lane; f < F; f += 32) atomicAdd(&ad[f], ex * hs[f]);
}

// out = relu(acc/denom + bias)
__global__ void gat_finish_relu(const float* __restrict__ acc, const float* __restrict__ denom,
                                const float* __restrict__ bias, float* __restrict__ out,
                                int N, int F) {
    long idx = (long)blockIdx.x * blockDim.x + threadIdx.x;
    if (idx >= (long)N * F) return;
    int dnode = (int)(idx / F), f = (int)(idx % F);
    float v = acc[idx] / denom[dnode] + bias[f];
    out[idx] = v > 0.f ? v : 0.f;
}

// final layer: out = (acc/denom + bias) row-L2-normalized; F == 32 == one lane/feature
__global__ void gat_finish_norm(const float* __restrict__ acc, const float* __restrict__ denom,
                                const float* __restrict__ bias, float* __restrict__ out, int N) {
    int w    = (blockIdx.x * blockDim.x + threadIdx.x) >> 5;
    int lane = threadIdx.x & 31;
    if (w >= N) return;
    float v = acc[(long)w * 32 + lane] / denom[w] + bias[lane];
    float sq = v * v;
#pragma unroll
    for (int off = 16; off > 0; off >>= 1) sq += __shfl_xor(sq, off, 32);
    float nrm = fmaxf(sqrtf(sq), 1e-12f);
    out[(long)w * 32 + lane] = v / nrm;
}

extern "C" void kernel_launch(void* const* d_in, const int* in_sizes, int n_in,
                              void* d_out, int out_size, void* d_ws, size_t ws_size,
                              hipStream_t stream) {
    const float*     x   = (const float*)d_in[0];
    const long long* ei  = (const long long*)d_in[1];   // int64 per reference
    const float*     Wp  = (const float*)d_in[2];
    const float*     bp  = (const float*)d_in[3];
    const float*     Wl[2]  = {(const float*)d_in[4],  (const float*)d_in[8]};
    const float*     asl[2] = {(const float*)d_in[5],  (const float*)d_in[9]};
    const float*     adl[2] = {(const float*)d_in[6],  (const float*)d_in[10]};
    const float*     bl[2]  = {(const float*)d_in[7],  (const float*)d_in[11]};
    const float*     W3  = (const float*)d_in[12];
    const float*     a3s = (const float*)d_in[13];
    const float*     a3d = (const float*)d_in[14];
    const float*     b3  = (const float*)d_in[15];

    const int Din = 128, Dh = 64;
    const int N = in_sizes[0] / Din;
    const int E = in_sizes[1] / 2;
    const int Etot = E + N;

    float*     buf0  = (float*)d_ws;                 // layer input   [N,64]
    float*     hbuf  = buf0 + (size_t)N * Dh;        // projection h  [N,64]
    float*     acc   = hbuf + (size_t)N * Dh;        // aggregation   [N,64]
    float*     esrc  = acc  + (size_t)N * Dh;        // [N]
    float*     edst  = esrc + N;                     // [N]
    unsigned*  mkey  = (unsigned*)(edst + N);        // [N]
    float*     denom = (float*)(mkey + N);           // [N]
    _Float16*  whalf = (_Float16*)(denom + N);       // [<= 64*128] f16 weights

    auto cdiv = [](long a, long b) { return (int)((a + b - 1) / b); };
    const int B = 256;
    const int gemmBlocks = cdiv(cdiv(N, 16), 8);     // 8 waves / 256-thread block

    // pre-linear: buf0 = x @ Wp^T + bp   (no attention dots)
    w_to_f16<<<cdiv(64 * 128, B), B, 0, stream>>>(Wp, whalf, 64 * 128);
    gat_gemm_wmma<128, 64><<<gemmBlocks, 256, 0, stream>>>(
        x, whalf, bp, nullptr, nullptr, buf0, nullptr, nullptr, N);

    // layers 1 & 2: 64 -> 64, ReLU
    for (int l = 0; l < 2; ++l) {
        w_to_f16<<<cdiv(64 * 64, B), B, 0, stream>>>(Wl[l], whalf, 64 * 64);
        gat_gemm_wmma<64, 64><<<gemmBlocks, 256, 0, stream>>>(
            buf0, whalf, nullptr, asl[l], adl[l], hbuf, esrc, edst, N);
        gat_init<<<cdiv((long)N * 64, B), B, 0, stream>>>(mkey, denom, acc, N, 64);
        gat_edge_max<<<cdiv(Etot, B), B, 0, stream>>>(ei, E, N, esrc, edst, mkey);
        gat_edge_acc<<<cdiv((long)Etot * 32, B), B, 0, stream>>>(ei, E, N, esrc, edst, mkey,
                                                                 hbuf, 64, denom, acc);
        gat_finish_relu<<<cdiv((long)N * 64, B), B, 0, stream>>>(acc, denom, bl[l], buf0, N, 64);
    }

    // layer 3: 64 -> 32, then row L2 normalize into d_out
    w_to_f16<<<cdiv(32 * 64, B), B, 0, stream>>>(W3, whalf, 32 * 64);
    gat_gemm_wmma<64, 32><<<gemmBlocks, 256, 0, stream>>>(
        buf0, whalf, nullptr, a3s, a3d, hbuf, esrc, edst, N);
    gat_init<<<cdiv((long)N * 64, B), B, 0, stream>>>(mkey, denom, acc, N, 32);
    gat_edge_max<<<cdiv(Etot, B), B, 0, stream>>>(ei, E, N, esrc, edst, mkey);
    gat_edge_acc<<<cdiv((long)Etot * 32, B), B, 0, stream>>>(ei, E, N, esrc, edst, mkey,
                                                             hbuf, 32, denom, acc);
    gat_finish_norm<<<cdiv((long)N * 32, B), B, 0, stream>>>(acc, denom, b3, (float*)d_out, N);
}
